// LargeScalePBGNNCrossInteraction_16758962389037
// MI455X (gfx1250) — matrix operand order repaired and checked by reading, not compiled, not run on hardware
//
#include <hip/hip_runtime.h>

typedef __attribute__((ext_vector_type(16))) _Float16 v16h;
typedef __attribute__((ext_vector_type(8)))  float    v8f;
typedef __attribute__((ext_vector_type(8)))  _Float16 h8v;
typedef __attribute__((ext_vector_type(4)))  _Float16 h4v;
typedef __attribute__((ext_vector_type(4)))  float    f4v;

// shifted softplus: log(1+e^x) - log(2), stable form with HW transcendentals
__device__ __forceinline__ float ssp_f(float v) {
    float e = __expf(-fabsf(v));                    // v_exp_f32
    return fmaxf(v, 0.0f) + __logf(1.0f + e) - 0.69314718055994531f;  // v_log_f32
}

__device__ __forceinline__ v8f wmma_16x16x32(v16h a, v16h b, v8f c) {
    return __builtin_amdgcn_wmma_f32_16x16x32_f16(false, a, false, b, (short)0, c, false, false);
}

// A-fragment (16-bit A 16x32): halves[0..7]=K k0..k0+7, halves[8..15]=K k0+16..k0+23
__device__ __forceinline__ v16h frag_from_f32(const float* p) {
    f4v x0 = *(const f4v*)(p);
    f4v x1 = *(const f4v*)(p + 4);
    f4v x2 = *(const f4v*)(p + 16);
    f4v x3 = *(const f4v*)(p + 20);
    v16h a;
#pragma unroll
    for (int t = 0; t < 4; ++t) {
        a[t]      = (_Float16)x0[t];
        a[4 + t]  = (_Float16)x1[t];
        a[8 + t]  = (_Float16)x2[t];
        a[12 + t] = (_Float16)x3[t];
    }
    return a;
}

// f16 source; lo 8 halves at p, hi 8 at p+gap (gap=16 for both A and B frags)
__device__ __forceinline__ v16h frag_from_h16(const _Float16* p, int gap) {
    h8v lo = *(const h8v*)(p);
    h8v hi = *(const h8v*)(p + gap);
    v16h a;
#pragma unroll
    for (int t = 0; t < 8; ++t) { a[t] = lo[t]; a[8 + t] = hi[t]; }
    return a;
}

// vector copy of pre-transposed f16 weights (global [rows][kp]) into padded LDS
__device__ __forceinline__ void stage_w(_Float16* lds, const _Float16* g,
                                        int rows, int kp, int ldstride,
                                        int tid, int nthr) {
    const int cpr = kp >> 3;                  // 8-half chunks per row
    for (int i = tid; i < rows * cpr; i += nthr) {
        int n = i / cpr, c = (i - n * cpr) * 8;
        *(h8v*)&lds[n * ldstride + c] = *(const h8v*)&g[n * kp + c];
    }
}

// ---------------------------------------------------------------------------
// Prep: transpose f32 weight [K][128] -> f16 Bt [128][Kp] (zero-padded K->Kp)
// ---------------------------------------------------------------------------
__global__ __launch_bounds__(256) void k_prep(const float* __restrict__ src,
                                              _Float16* __restrict__ dst,
                                              int K, int Kp) {
    int i = blockIdx.x * 256 + threadIdx.x;    // over 128*Kp
    if (i >= 128 * Kp) return;
    int n = i / Kp, k = i - n * Kp;
    dst[i] = (k < K) ? (_Float16)src[k * 128 + n] : (_Float16)0.0f;
}

// ---------------------------------------------------------------------------
// Kernel 1: node projections xf = x @ W, yf = y @ W_y (f16 out)
// 256 threads (8 waves), TPW tiles per wave; gridDim.y selects stream
// ---------------------------------------------------------------------------
#define TPW_P 4
__global__ __launch_bounds__(256) void k_project(
    const float* __restrict__ X0, const float* __restrict__ X1,
    const _Float16* __restrict__ Bt0, const _Float16* __restrict__ Bt1,
    _Float16* __restrict__ O0, _Float16* __restrict__ O1, int N)
{
    __shared__ _Float16 sBt[128 * 136];

    const float*    X  = (blockIdx.y == 0) ? X0 : X1;
    const _Float16* Bt = (blockIdx.y == 0) ? Bt0 : Bt1;
    _Float16*       O  = (blockIdx.y == 0) ? O0 : O1;

    const int tid = threadIdx.x;
    stage_w(sBt, Bt, 128, 128, 136, tid, 256);
    __syncthreads();

    const int lane = tid & 31, wave = tid >> 5;
    const int grp = lane >> 4, r = lane & 15;
    long tbase = ((long)blockIdx.x * 8 + wave) * TPW_P;

    for (int t = 0; t < TPW_P; ++t) {
        long row0 = (tbase + t) * 16;
        if (row0 >= N) break;
        const bool full = (row0 + 16 <= N);
        long row = row0 + r; if (row >= N) row = N - 1;

        v16h af[4];
#pragma unroll
        for (int ks = 0; ks < 4; ++ks)
            af[ks] = frag_from_f32(X + row * 128 + ks * 32 + grp * 8);

#pragma unroll
        for (int nt = 0; nt < 8; ++nt) {
            v16h bfr[4];
#pragma unroll
            for (int ks = 0; ks < 4; ++ks)
                bfr[ks] = frag_from_h16(&sBt[(nt * 16 + r) * 136 + ks * 32 + grp * 8], 16);
            v8f acc = {};
#pragma unroll
            for (int ks = 0; ks < 4; ++ks)
                acc = wmma_16x16x32(af[ks], bfr[ks], acc);

            int col = nt * 16 + r;
            long ob = row0 + grp * 8;
            if (full) {
#pragma unroll
                for (int v = 0; v < 8; ++v)
                    O[(ob + v) * 128 + col] = (_Float16)acc[v];
            } else {
#pragma unroll
                for (int v = 0; v < 8; ++v)
                    if (ob + v < N) O[(ob + v) * 128 + col] = (_Float16)acc[v];
            }
        }
    }
}

// ---------------------------------------------------------------------------
// Kernel 2 (fused): edge filter MLP + gather-multiply + scatter-add atomics
// 128 threads (4 waves), TPW tiles per wave
// ---------------------------------------------------------------------------
#define TPW_E 4
__global__ __launch_bounds__(128) void k_edges(
    const float* __restrict__ f_ij, const float* __restrict__ rcut,
    const int* __restrict__ idx_i, const int* __restrict__ idx_j,
    const _Float16* __restrict__ Bt1, const float* __restrict__ bf1,
    const _Float16* __restrict__ Bt2, const float* __restrict__ bf2,
    const _Float16* __restrict__ xfh, const _Float16* __restrict__ yfh,
    float* __restrict__ conv_x, float* __restrict__ conv_y, int E)
{
    __shared__ _Float16 sW1[128 * 40];        // Wf1^T f16, K padded 20->32
    __shared__ _Float16 sW2[128 * 136];       // Wf2^T f16
    __shared__ _Float16 sStg[4 * 16 * 136];   // per-wave 16x128 staging (f16)

    const int tid = threadIdx.x;
    stage_w(sW1, Bt1, 128, 32, 40, tid, 128);
    stage_w(sW2, Bt2, 128, 128, 136, tid, 128);
    __syncthreads();

    const int lane = tid & 31, wave = tid >> 5;
    const int grp = lane >> 4, r = lane & 15;
    _Float16* stg = &sStg[wave * 16 * 136];
    const int c = lane * 4;                   // scatter columns
    long tbase = ((long)blockIdx.x * 4 + wave) * TPW_E;

    for (int t = 0; t < TPW_E; ++t) {
        long e0 = (tbase + t) * 16;
        if (e0 >= E) break;
        const bool full = (e0 + 16 <= E);
        long erow = e0 + r; if (erow >= E) erow = E - 1;

        // A fragment of f_ij (row stride R=20), zero-padded K 20->32
        v16h afe;
        {
            const float* p = f_ij + erow * 20 + grp * 8;
#pragma unroll
            for (int tt = 0; tt < 8; ++tt) afe[tt] = (_Float16)p[tt];
#pragma unroll
            for (int tt = 0; tt < 8; ++tt) {
                int k = grp * 8 + 16 + tt;
                afe[8 + tt] = (k < 20) ? (_Float16)p[16 + tt] : (_Float16)0.0f;
            }
        }

        // GEMM1: ssp(f_ij @ Wf1 + bf1) -> staging
#pragma unroll
        for (int nt = 0; nt < 8; ++nt) {
            v16h b = frag_from_h16(&sW1[(nt * 16 + r) * 40 + grp * 8], 16);
            v8f acc = {};
            acc = wmma_16x16x32(afe, b, acc);
            int col = nt * 16 + r;
            float bb = bf1[col];
#pragma unroll
            for (int v = 0; v < 8; ++v)
                stg[(grp * 8 + v) * 136 + col] = (_Float16)ssp_f(acc[v] + bb);
        }

        // reload staging as A fragments (per-wave DS ops are in-order)
        v16h ah[4];
#pragma unroll
        for (int ks = 0; ks < 4; ++ks)
            ah[ks] = frag_from_h16(&stg[r * 136 + ks * 32 + grp * 8], 16);

        float rc[8];
#pragma unroll
        for (int v = 0; v < 8; ++v) {
            long e = e0 + grp * 8 + v;
            rc[v] = (e < E) ? rcut[e] : 0.0f;
        }

        // GEMM2: Wij = (h @ Wf2 + bf2) * rcut -> staging (overwrite)
#pragma unroll
        for (int nt = 0; nt < 8; ++nt) {
            v16h bfr[4];
#pragma unroll
            for (int ks = 0; ks < 4; ++ks)
                bfr[ks] = frag_from_h16(&sW2[(nt * 16 + r) * 136 + ks * 32 + grp * 8], 16);
            v8f acc = {};
#pragma unroll
            for (int ks = 0; ks < 4; ++ks)
                acc = wmma_16x16x32(ah[ks], bfr[ks], acc);

            int col = nt * 16 + r;
            float bb = bf2[col];
#pragma unroll
            for (int v = 0; v < 8; ++v)
                stg[(grp * 8 + v) * 136 + col] = (_Float16)((acc[v] + bb) * rc[v]);
        }

        // gather-multiply-scatter: 32 lanes per edge, 4 columns each
        int mmax = full ? 16 : (int)(E - e0);
        for (int m = 0; m < mmax; ++m) {
            long e = e0 + m;
            int i = idx_i[e], j = idx_j[e];
            h4v wh = *(const h4v*)&stg[m * 136 + c];
            h4v yv = *(const h4v*)(yfh + (long)j * 128 + c);
            h4v xv = *(const h4v*)(xfh + (long)i * 128 + c);
            float* px = conv_x + (long)i * 128 + c;
            float* py = conv_y + (long)j * 128 + c;
#pragma unroll
            for (int q = 0; q < 4; ++q) {
                float w = (float)wh[q];
                unsafeAtomicAdd(px + q, (float)yv[q] * w);
                unsafeAtomicAdd(py + q, (float)xv[q] * w);
            }
        }
    }
}

// ---------------------------------------------------------------------------
// Kernel 3: out = ssp(conv @ W1 + b1) @ W2 + b2 (weights phased through LDS)
// 128 threads (4 waves); gridDim.y selects stream
// ---------------------------------------------------------------------------
__global__ __launch_bounds__(128) void k_out(
    const float* __restrict__ conv_x, const float* __restrict__ conv_y,
    const _Float16* __restrict__ Btx1, const float* __restrict__ bx1,
    const _Float16* __restrict__ Btx2, const float* __restrict__ bx2,
    const _Float16* __restrict__ Bty1, const float* __restrict__ by1,
    const _Float16* __restrict__ Bty2, const float* __restrict__ by2,
    float* __restrict__ out, int N)
{
    __shared__ _Float16 sW[128 * 136];        // reused: W1 then W2
    __shared__ _Float16 sStg[4 * 16 * 136];

    const bool isx = (blockIdx.y == 0);
    const float*    C   = isx ? conv_x : conv_y;
    const _Float16* B1  = isx ? Btx1 : Bty1;
    const float*    b1  = isx ? bx1 : by1;
    const _Float16* B2  = isx ? Btx2 : Bty2;
    const float*    b2  = isx ? bx2 : by2;
    float* O = out + (isx ? 0 : (long)N * 128);

    const int tid = threadIdx.x;
    const int lane = tid & 31, wave = tid >> 5;
    const int grp = lane >> 4, r = lane & 15;
    _Float16* stg = &sStg[wave * 16 * 136];

    long row0 = ((long)blockIdx.x * 4 + wave) * 16;
    const bool valid = (row0 < N);
    const bool full  = (row0 + 16 <= N);
    long row = row0 + r; if (row >= N) row = (N > 0) ? N - 1 : 0;

    // phase 1: W1
    stage_w(sW, B1, 128, 128, 136, tid, 128);
    __syncthreads();

    if (valid) {
        v16h a1[4];
#pragma unroll
        for (int ks = 0; ks < 4; ++ks)
            a1[ks] = frag_from_f32(C + row * 128 + ks * 32 + grp * 8);
#pragma unroll
        for (int nt = 0; nt < 8; ++nt) {
            v16h bfr[4];
#pragma unroll
            for (int ks = 0; ks < 4; ++ks)
                bfr[ks] = frag_from_h16(&sW[(nt * 16 + r) * 136 + ks * 32 + grp * 8], 16);
            v8f acc = {};
#pragma unroll
            for (int ks = 0; ks < 4; ++ks)
                acc = wmma_16x16x32(a1[ks], bfr[ks], acc);

            int col = nt * 16 + r;
            float bb = b1[col];
#pragma unroll
            for (int v = 0; v < 8; ++v)
                stg[(grp * 8 + v) * 136 + col] = (_Float16)ssp_f(acc[v] + bb);
        }
    }
    __syncthreads();

    // phase 2: W2 (overwrites sW; staging preserved)
    stage_w(sW, B2, 128, 128, 136, tid, 128);
    __syncthreads();

    if (valid) {
        v16h a2[4];
#pragma unroll
        for (int ks = 0; ks < 4; ++ks)
            a2[ks] = frag_from_h16(&stg[r * 136 + ks * 32 + grp * 8], 16);
#pragma unroll
        for (int nt = 0; nt < 8; ++nt) {
            v16h bfr[4];
#pragma unroll
            for (int ks = 0; ks < 4; ++ks)
                bfr[ks] = frag_from_h16(&sW[(nt * 16 + r) * 136 + ks * 32 + grp * 8], 16);
            v8f acc = {};
#pragma unroll
            for (int ks = 0; ks < 4; ++ks)
                acc = wmma_16x16x32(a2[ks], bfr[ks], acc);

            int col = nt * 16 + r;
            float bb = b2[col];
            long ob = row0 + grp * 8;
            if (full) {
#pragma unroll
                for (int v = 0; v < 8; ++v)
                    O[(ob + v) * 128 + col] = acc[v] + bb;
            } else {
#pragma unroll
                for (int v = 0; v < 8; ++v)
                    if (ob + v < N) O[(ob + v) * 128 + col] = acc[v] + bb;
            }
        }
    }
}

// ---------------------------------------------------------------------------
extern "C" void kernel_launch(void* const* d_in, const int* in_sizes, int n_in,
                              void* d_out, int out_size, void* d_ws, size_t ws_size,
                              hipStream_t stream)
{
    const float* x       = (const float*)d_in[0];
    const float* y       = (const float*)d_in[1];
    const float* f_ij    = (const float*)d_in[2];
    const float* rcut    = (const float*)d_in[3];
    const int*   idx_i   = (const int*)d_in[4];
    const int*   idx_j   = (const int*)d_in[5];
    const float* W_in2f  = (const float*)d_in[6];
    const float* W_in2fy = (const float*)d_in[7];
    const float* Wf1 = (const float*)d_in[8];   const float* bf1 = (const float*)d_in[9];
    const float* Wf2 = (const float*)d_in[10];  const float* bf2 = (const float*)d_in[11];
    const float* Wx1 = (const float*)d_in[12];  const float* bx1 = (const float*)d_in[13];
    const float* Wx2 = (const float*)d_in[14];  const float* bx2 = (const float*)d_in[15];
    const float* Wy1 = (const float*)d_in[16];  const float* by1 = (const float*)d_in[17];
    const float* Wy2 = (const float*)d_in[18];  const float* by2 = (const float*)d_in[19];

    const int N = in_sizes[0] / 128;
    const int E = in_sizes[3];

    // workspace layout
    char* ws = (char*)d_ws;
    size_t o = 0;
    _Float16* xfh = (_Float16*)(ws + o); o += (size_t)N * 128 * 2;
    _Float16* yfh = (_Float16*)(ws + o); o += (size_t)N * 128 * 2;
    float*    cvx = (float*)(ws + o);    o += (size_t)N * 128 * 4;
    float*    cvy = (float*)(ws + o);    o += (size_t)N * 128 * 4;
    _Float16* btA = (_Float16*)(ws + o); o += 16384 * 2;   // W_in2f^T
    _Float16* btB = (_Float16*)(ws + o); o += 16384 * 2;   // W_in2f_y^T
    _Float16* bt1 = (_Float16*)(ws + o); o += 4096 * 2;    // Wf1^T (Kp=32)
    _Float16* bt2 = (_Float16*)(ws + o); o += 16384 * 2;   // Wf2^T
    _Float16* btx1 = (_Float16*)(ws + o); o += 16384 * 2;
    _Float16* btx2 = (_Float16*)(ws + o); o += 16384 * 2;
    _Float16* bty1 = (_Float16*)(ws + o); o += 16384 * 2;
    _Float16* bty2 = (_Float16*)(ws + o); o += 16384 * 2;

    hipMemsetAsync(cvx, 0, (size_t)N * 128 * 8, stream);   // conv_x + conv_y

    // transpose/convert weights once
    k_prep<<<64, 256, 0, stream>>>(W_in2f,  btA, 128, 128);
    k_prep<<<64, 256, 0, stream>>>(W_in2fy, btB, 128, 128);
    k_prep<<<16, 256, 0, stream>>>(Wf1,     bt1,  20,  32);
    k_prep<<<64, 256, 0, stream>>>(Wf2,     bt2, 128, 128);
    k_prep<<<64, 256, 0, stream>>>(Wx1,     btx1, 128, 128);
    k_prep<<<64, 256, 0, stream>>>(Wx2,     btx2, 128, 128);
    k_prep<<<64, 256, 0, stream>>>(Wy1,     bty1, 128, 128);
    k_prep<<<64, 256, 0, stream>>>(Wy2,     bty2, 128, 128);

    const int tilesN = (N + 15) / 16;
    const int tilesE = (E + 15) / 16;

    k_project<<<dim3((tilesN + 8 * TPW_P - 1) / (8 * TPW_P), 2), 256, 0, stream>>>(
        x, y, btA, btB, xfh, yfh, N);

    k_edges<<<dim3((tilesE + 4 * TPW_E - 1) / (4 * TPW_E)), 128, 0, stream>>>(
        f_ij, rcut, idx_i, idx_j, bt1, bf1, bt2, bf2, xfh, yfh, cvx, cvy, E);

    k_out<<<dim3((tilesN + 3) / 4, 2), 128, 0, stream>>>(
        cvx, cvy, btx1, bx1, btx2, bx2, bty1, by1, bty2, by2, (float*)d_out, N);
}
